// sLSTMLayer_29111288333271
// MI455X (gfx1250) — matrix operand order repaired and checked by reading, not compiled, not run on hardware
//
#include <hip/hip_runtime.h>

// ---------------- problem constants ----------------
#define NBLK 32     // persistent workgroups; each owns 32 of the 1024 gate cols
#define TPB  256    // 8 wave32 waves
#define S_   2048
#define B_   64
#define D_   256
#define H_   256
#define FH_  1024   // 4*H
#define L_   2

typedef __attribute__((ext_vector_type(16))) __bf16 v16bf;
typedef __attribute__((ext_vector_type(8)))  float  v8f;

__device__ inline __bf16 f2bf(float f) {
  unsigned u = __float_as_uint(f);
  unsigned r = (u + 0x7FFFu + ((u >> 16) & 1u)) >> 16;   // round-to-nearest-even
  unsigned short s = (unsigned short)r;
  __bf16 b;
  __builtin_memcpy(&b, &s, 2);
  return b;
}

// Monotonic grid barrier: counter in ws, zeroed by hipMemsetAsync before launch.
__device__ inline void gbar(unsigned* cnt, unsigned target) {
  __syncthreads();
  if (threadIdx.x == 0) {
    __threadfence();
    __hip_atomic_fetch_add(cnt, 1u, __ATOMIC_ACQ_REL, __HIP_MEMORY_SCOPE_AGENT);
    while (__hip_atomic_load(cnt, __ATOMIC_ACQUIRE, __HIP_MEMORY_SCOPE_AGENT) < target)
      __builtin_amdgcn_s_sleep(2);
  }
  __syncthreads();
  __threadfence();   // invalidate stale near-cache lines for all lanes
}

// Block-wide sum (256 threads), result broadcast. Wave32 shuffle + LDS.
__device__ inline float bsum(float v, float* red) {
  #pragma unroll
  for (int o = 16; o > 0; o >>= 1) v += __shfl_xor(v, o, 32);
  int w = threadIdx.x >> 5;
  __syncthreads();
  if ((threadIdx.x & 31) == 0) red[w] = v;
  __syncthreads();
  float t = 0.f;
  #pragma unroll
  for (int i = 0; i < 8; i++) t += red[i];
  return t;
}

__global__ void __launch_bounds__(TPB)
slstm_persistent(const float* __restrict__ x,
                 const float* __restrict__ Wx,  const float* __restrict__ bx,
                 const float* __restrict__ Wh,  const float* __restrict__ bhv,
                 const float* __restrict__ g_in, const float* __restrict__ b_in,
                 const float* __restrict__ g_hu, const float* __restrict__ b_hu,
                 const float* __restrict__ g_c,  const float* __restrict__ b_c,
                 const float* __restrict__ g_h,  const float* __restrict__ b_h,
                 const float* __restrict__ g_ln, const float* __restrict__ b_ln,
                 float* __restrict__ out, float* __restrict__ ws)
{
  extern __shared__ char smem[];
  __bf16* lWx = (__bf16*)smem;               // [L][32][256]  bf16  (32 KB)
  __bf16* lWh = lWx + L_ * 32 * D_;          // [L][32][256]  bf16  (32 KB)
  __bf16* lA  = lWh + L_ * 32 * D_;          // layer_in [64][256] bf16 (32 KB)
  __bf16* lH  = lA  + B_ * D_;               // h_prev   [64][256] bf16 (32 KB)
  float*  red = (float*)(lH + B_ * D_);      // reduction scratch (8 waves)

  const unsigned blk  = blockIdx.x;
  const unsigned tid  = threadIdx.x;
  const unsigned lane = tid & 31u;
  const unsigned wv   = tid >> 5;            // wave id 0..7
  const unsigned half = lane >> 4;           // lane half (ISA fragment layouts)
  const unsigned l16  = lane & 15u;

  unsigned* bar = (unsigned*)ws;
  float* gxr    = ws + 64;                   // raw x-gates  [64][1024]
  float* ghr    = gxr + B_ * FH_;            // raw h-gates  [64][1024]
  float* hstate = ghr + B_ * FH_;            // h [L][64][256] f32
  float* cstate = hstate + L_ * B_ * H_;     // c [L][64][256] f32
  __bf16* hbf   = (__bf16*)(cstate + L_ * B_ * H_); // bf16 shadow of h [L][64][256]

  // ---- one-time: stage this block's weight slices into LDS (f32 -> bf16) ----
  for (int l = 0; l < L_; l++)
    for (int idx = tid; idx < 32 * D_; idx += TPB) {
      int n = idx >> 8, k = idx & (D_ - 1);
      lWx[(l * 32 + n) * D_ + k] = f2bf(Wx[((size_t)l * FH_ + blk * 32 + n) * D_ + k]);
      lWh[(l * 32 + n) * D_ + k] = f2bf(Wh[((size_t)l * FH_ + blk * 32 + n) * D_ + k]);
    }
  // ---- zero recurrent state (deterministic per launch) ----
  for (int i = tid + blk * TPB; i < L_ * B_ * H_; i += NBLK * TPB) {
    hstate[i] = 0.f; cstate[i] = 0.f;
  }
  {
    unsigned* hb32 = (unsigned*)hbf;
    for (int i = tid + blk * TPB; i < L_ * B_ * H_ / 2; i += NBLK * TPB) hb32[i] = 0u;
  }
  unsigned gen = 1;
  gbar(bar, gen * NBLK);

  // wave tile assignment: 4 M-tiles x 2 N-tiles = 8 waves
  const unsigned m0  = (wv >> 1) * 16;               // batch-row tile base
  const unsigned nl0 = (wv & 1) * 16;                // local col tile base
  const unsigned ng0 = blk * 32 + nl0;               // global gate col base

  for (int t = 0; t < S_; t++) {
    const float* xt = x + (size_t)t * B_ * D_;
    for (int l = 0; l < L_; l++) {
      // ---- stage layer_in and h_prev[l] into LDS (bf16) ----
      if (l == 0) {
        // layer 0 input: f32 x[t] -> bf16 LDS; also prefetch x[t+1]
        for (int i = tid; i < B_ * D_; i += TPB) lA[i] = f2bf(xt[i]);
        if (t + 1 < S_) {
          const float* xn = xt + B_ * D_;
          int idx = (int)(blk * TPB + tid) * 16;     // one line per thread
          if (idx < B_ * D_) __builtin_prefetch(xn + idx, 0, 1);
        }
      } else {
        // layer 1 input = layer-0 fresh h, already bf16 in shadow buffer
        const unsigned* src = (const unsigned*)hbf;  // layer 0 slice
        unsigned* dst = (unsigned*)lA;
        for (int i = tid; i < B_ * D_ / 2; i += TPB) dst[i] = src[i];
      }
      {
        const unsigned* src = (const unsigned*)(hbf + (size_t)l * B_ * H_);
        unsigned* dst = (unsigned*)lH;
        for (int i = tid; i < B_ * D_ / 2; i += TPB) dst[i] = src[i];
      }
      __syncthreads();

      // ================= Phase A: dual WMMA GEMM slice =================
      v8f accx = {}, acch = {};
      #pragma unroll
      for (int kk = 0; kk < D_ / 32; kk++) {
        const int kb = kk * 32;
        v16bf af, hf, bxf, bhf;
        // A fragments (16x32 bf16): lanes 0-15 = M rows, K pairs in VGPRs,
        // K+8/K+24 halves in lanes 16-31 (ISA 7.12.2)
        #pragma unroll
        for (int v = 0; v < 8; v++) {
          const int k = kb + (v >> 2) * 16 + half * 8 + (v & 3) * 2;
          const __bf16* pa = &lA[(m0 + l16) * D_ + k];
          const __bf16* ph = &lH[(m0 + l16) * D_ + k];
          af[2 * v] = pa[0]; af[2 * v + 1] = pa[1];
          hf[2 * v] = ph[0]; hf[2 * v + 1] = ph[1];
        }
        // B fragments (32x16 bf16): N in lanes, K pairs per VGPR, same K order as A
        #pragma unroll
        for (int v = 0; v < 8; v++) {
          const int k = kb + (v >> 2) * 16 + half * 8 + (v & 3) * 2;
          const __bf16* pb = &lWx[((unsigned)l * 32 + nl0 + l16) * D_ + k];
          const __bf16* pc = &lWh[((unsigned)l * 32 + nl0 + l16) * D_ + k];
          bxf[2 * v] = pb[0]; bxf[2 * v + 1] = pb[1];
          bhf[2 * v] = pc[0]; bhf[2 * v + 1] = pc[1];
        }
        accx = __builtin_amdgcn_wmma_f32_16x16x32_bf16(false, af, false, bxf,
                                                       (short)0, accx, false, false);
        acch = __builtin_amdgcn_wmma_f32_16x16x32_bf16(false, hf, false, bhf,
                                                       (short)0, acch, false, false);
      }
      // D layout: lane = N col, VGPR r -> M = r + 8*(lane/16). Add bias, spill raw.
      #pragma unroll
      for (int r = 0; r < 8; r++) {
        const int m = m0 + half * 8 + r;
        const int n = ng0 + l16;
        gxr[m * FH_ + n] = accx[r] + bx[l * FH_ + n];
        ghr[m * FH_ + n] = acch[r] + bhv[l * FH_ + n];
      }
      gen++; gbar(bar, gen * NBLK);

      // ============ Phase B: LN + gates + cell update (2 rows/block) ============
      for (int rr = 0; rr < 2; rr++) {
        const int row = blk * 2 + rr;
        // row stats over 4H=1024 for gx and gh
        float sx = 0.f, sxx = 0.f, sh = 0.f, shh = 0.f;
        for (int n = tid; n < FH_; n += TPB) {
          float a = gxr[row * FH_ + n], b = ghr[row * FH_ + n];
          sx += a; sxx += a * a; sh += b; shh += b * b;
        }
        sx  = bsum(sx, red);  sxx = bsum(sxx, red);
        sh  = bsum(sh, red);  shh = bsum(shh, red);
        const float mx = sx * (1.f / FH_), vx = sxx * (1.f / FH_) - mx * mx;
        const float mh = sh * (1.f / FH_), vh = shh * (1.f / FH_) - mh * mh;
        const float rx = rsqrtf(vx + 1e-5f), rh = rsqrtf(vh + 1e-5f);

        const int j = tid;                           // 256 threads == H
        auto gate = [&](int n) -> float {
          float a = g_in[l * FH_ + n] * (gxr[row * FH_ + n] - mx) * rx + b_in[l * FH_ + n];
          float b = g_hu[l * FH_ + n] * (ghr[row * FH_ + n] - mh) * rh + b_hu[l * FH_ + n];
          return a + b;
        };
        const float si = gate(j), sf = gate(H_ + j), sg = gate(2 * H_ + j), so = gate(3 * H_ + j);
        const float ig = __expf(fminf(fmaxf(si, -5.f), 5.f));
        const float fg = __expf(fminf(fmaxf(sf, -5.f), 5.f));
        const float gg = tanhf(sg);
        const float og = 1.f / (1.f + __expf(-so));

        // cell update + LN over H
        float craw = fg * cstate[((size_t)l * B_ + row) * H_ + j] + ig * gg;
        float cs = bsum(craw, red), css = bsum(craw * craw, red);
        float mc = cs * (1.f / H_), vc = css * (1.f / H_) - mc * mc;
        float cl = g_c[l * H_ + j] * (craw - mc) * rsqrtf(vc + 1e-5f) + b_c[l * H_ + j];
        cstate[((size_t)l * B_ + row) * H_ + j] = cl;

        // hidden: LN(o*tanh(cl)) + residual, then final LN
        float hraw = og * tanhf(cl);
        float hs = bsum(hraw, red), hss = bsum(hraw * hraw, red);
        float mhd = hs * (1.f / H_), vhd = hss * (1.f / H_) - mhd * mhd;
        float hl1 = g_h[l * H_ + j] * (hraw - mhd) * rsqrtf(vhd + 1e-5f) + b_h[l * H_ + j];
        float resid = (l == 0) ? xt[row * D_ + j] : hstate[(size_t)row * H_ + j];
        float h2 = hl1 + resid;
        float h2s = bsum(h2, red), h2ss = bsum(h2 * h2, red);
        float m2 = h2s * (1.f / H_), v2 = h2ss * (1.f / H_) - m2 * m2;
        float h3 = g_ln[l * H_ + j] * (h2 - m2) * rsqrtf(v2 + 1e-5f) + b_ln[l * H_ + j];
        hstate[((size_t)l * B_ + row) * H_ + j] = h3;
        hbf[((size_t)l * B_ + row) * H_ + j] = f2bf(h3);   // bf16 shadow for next GEMM
        if (l == L_ - 1) out[((size_t)t * B_ + row) * H_ + j] = h3;
      }
      gen++; gbar(bar, gen * NBLK);
    }
  }

  // final (h, c) -> tail of d_out
  float* outh = out + (size_t)S_ * B_ * H_;
  float* outc = outh + (size_t)L_ * B_ * H_;
  for (int i = tid + blk * TPB; i < L_ * B_ * H_; i += NBLK * TPB) {
    outh[i] = hstate[i];
    outc[i] = cstate[i];
  }
}

extern "C" void kernel_launch(void* const* d_in, const int* in_sizes, int n_in,
                              void* d_out, int out_size, void* d_ws, size_t ws_size,
                              hipStream_t stream) {
  const float* x    = (const float*)d_in[0];
  const float* Wx   = (const float*)d_in[1];
  const float* bx   = (const float*)d_in[2];
  const float* Wh   = (const float*)d_in[3];
  const float* bh   = (const float*)d_in[4];
  const float* g_in = (const float*)d_in[5];
  const float* b_in = (const float*)d_in[6];
  const float* g_hu = (const float*)d_in[7];
  const float* b_hu = (const float*)d_in[8];
  const float* g_c  = (const float*)d_in[9];
  const float* b_c  = (const float*)d_in[10];
  const float* g_h  = (const float*)d_in[11];
  const float* b_h  = (const float*)d_in[12];
  const float* g_ln = (const float*)d_in[13];
  const float* b_ln = (const float*)d_in[14];
  float* out = (float*)d_out;
  float* ws  = (float*)d_ws;

  // zero the grid-barrier counter region (graph-capture safe, stream-ordered)
  hipMemsetAsync(d_ws, 0, 256, stream);

  // dynamic LDS: lWx+lWh (64KB) + lA+lH (64KB) + reduction scratch
  size_t smem = ((size_t)(2 * L_ * 32 * D_) + (size_t)(2 * B_ * D_)) * 2 + 64 * sizeof(float);

  slstm_persistent<<<dim3(NBLK), dim3(TPB), smem, stream>>>(
      x, Wx, bx, Wh, bh, g_in, b_in, g_hu, b_hu,
      g_c, b_c, g_h, b_h, g_ln, b_ln, out, ws);
}
